// GATv2Layer_18786186952938
// MI455X (gfx1250) — matrix operand order repaired and checked by reading, not compiled, not run on hardware
//
#include <hip/hip_runtime.h>

typedef __attribute__((ext_vector_type(16))) _Float16 v16h;
typedef __attribute__((ext_vector_type(8)))  float    v8f;

#define NB      2
#define NHEADS  4
#define HDIM    32
#define DMODEL  128
#define NNODES  4096
#define NEG_SLOPE 0.2f
#define LOG2E   1.4426950408889634f
#define LN_EPS  1e-5f

// ---------------------------------------------------------------------------
// Kernel 1: h = x @ W via WMMA (f16 in, f32 acc).
//   Writes hT (b, d, j) in f16 (j contiguous -> WMMA B frags are 32B loads),
//   and per-node scores s = h.att_src, t = h.att_dst (f32).
// Block: 128 threads (4 waves). Wave w computes output cols [32w, 32w+32)
// (== head w's dims). Each block handles 16 rows of the 8192-row batch.
// ---------------------------------------------------------------------------
__global__ __launch_bounds__(128) void k_proj(
    const float* __restrict__ x, const float* __restrict__ W,
    const float* __restrict__ att_src, const float* __restrict__ att_dst,
    _Float16* __restrict__ hT, float* __restrict__ sv, float* __restrict__ tv)
{
    __shared__ __align__(32) _Float16 WTs[DMODEL * 136];  // W^T, f16, padded stride

    const int tid  = threadIdx.x;
    const int lane = tid & 31;
    const int wave = tid >> 5;
    const int mrow = lane & 15;
    const int g    = lane >> 4;
    const int r0   = blockIdx.x * 16;          // global row base in [0, 8192)

    // Stage W transposed into LDS as f16: WTs[n*136 + k]. Thread t owns k-row t.
    {
        const float4* wrow = reinterpret_cast<const float4*>(W + (size_t)tid * DMODEL);
#pragma unroll
        for (int n4 = 0; n4 < DMODEL / 4; ++n4) {
            float4 v = wrow[n4];
            WTs[(n4 * 4 + 0) * 136 + tid] = (_Float16)v.x;
            WTs[(n4 * 4 + 1) * 136 + tid] = (_Float16)v.y;
            WTs[(n4 * 4 + 2) * 136 + tid] = (_Float16)v.z;
            WTs[(n4 * 4 + 3) * 136 + tid] = (_Float16)v.w;
        }
    }
    __syncthreads();

    const int nbase = wave * 32;
    v8f acc0 = {}, acc1 = {};
    const float* xrow = x + (size_t)(r0 + mrow) * DMODEL;

#pragma unroll
    for (int ks = 0; ks < 4; ++ks) {
        const int kb = ks * 32;
        // A fragment: lane holds row mrow, K = kb+16g..+15
        float4 a0 = *reinterpret_cast<const float4*>(xrow + kb + 16 * g);
        float4 a1 = *reinterpret_cast<const float4*>(xrow + kb + 16 * g + 4);
        float4 a2 = *reinterpret_cast<const float4*>(xrow + kb + 16 * g + 8);
        float4 a3 = *reinterpret_cast<const float4*>(xrow + kb + 16 * g + 12);
        v16h A;
        A[0] = (_Float16)a0.x; A[1] = (_Float16)a0.y; A[2] = (_Float16)a0.z; A[3] = (_Float16)a0.w;
        A[4] = (_Float16)a1.x; A[5] = (_Float16)a1.y; A[6] = (_Float16)a1.z; A[7] = (_Float16)a1.w;
        A[8] = (_Float16)a2.x; A[9] = (_Float16)a2.y; A[10] = (_Float16)a2.z; A[11] = (_Float16)a2.w;
        A[12] = (_Float16)a3.x; A[13] = (_Float16)a3.y; A[14] = (_Float16)a3.z; A[15] = (_Float16)a3.w;

        // B fragments: lane holds col n = nbase + nt*16 + mrow, K = kb+16g..+15
        const v16h B0 = *reinterpret_cast<const v16h*>(&WTs[(nbase + mrow) * 136 + kb + 16 * g]);
        const v16h B1 = *reinterpret_cast<const v16h*>(&WTs[(nbase + 16 + mrow) * 136 + kb + 16 * g]);

        acc0 = __builtin_amdgcn_wmma_f32_16x16x32_f16(false, A, false, B0, (short)0, acc0, false, false);
        acc1 = __builtin_amdgcn_wmma_f32_16x16x32_f16(false, A, false, B1, (short)0, acc1, false, false);
    }

    // Per-head scores: s_i = sum_d h[i,d]*att_src[w,d]. Acc layout: lane holds
    // col n = tile_base + mrow, rows M = 8g + r (r = VGPR index).
    const float as0 = att_src[wave * HDIM + mrow];
    const float as1 = att_src[wave * HDIM + 16 + mrow];
    const float ad0 = att_dst[wave * HDIM + mrow];
    const float ad1 = att_dst[wave * HDIM + 16 + mrow];
    float sred[8], tred[8];
#pragma unroll
    for (int r = 0; r < 8; ++r) {
        sred[r] = acc0[r] * as0 + acc1[r] * as1;
        tred[r] = acc0[r] * ad0 + acc1[r] * ad1;
    }
#pragma unroll
    for (int m = 1; m < 16; m <<= 1) {
#pragma unroll
        for (int r = 0; r < 8; ++r) {
            sred[r] += __shfl_xor(sred[r], m, 32);
            tred[r] += __shfl_xor(tred[r], m, 32);
        }
    }
    const int b  = r0 >> 12;                 // /4096
    const int ib = (r0 & 4095) + 8 * g;      // row-in-batch for this lane group
    if (mrow == 0) {
        float* sp = sv + (size_t)(b * NHEADS + wave) * NNODES + ib;
        float* tp = tv + (size_t)(b * NHEADS + wave) * NNODES + ib;
#pragma unroll
        for (int r = 0; r < 8; ++r) { sp[r] = sred[r]; tp[r] = tred[r]; }
    }

    // Write hT[b][d][i] in f16: per lane pack 8 consecutive-row values (16B).
#pragma unroll
    for (int nt = 0; nt < 2; ++nt) {
        const int d = nbase + nt * 16 + mrow;
        union { _Float16 h[8]; uint4 u; } pk;
#pragma unroll
        for (int r = 0; r < 8; ++r) pk.h[r] = (_Float16)((nt == 0) ? acc0[r] : acc1[r]);
        *reinterpret_cast<uint4*>(hT + ((size_t)(b * DMODEL + d) * NNODES + ib)) = pk.u;
    }
}

// ---------------------------------------------------------------------------
// Kernel 1b: per-(b,head) max of t  (upper bound for softmax stabilization)
// ---------------------------------------------------------------------------
__global__ __launch_bounds__(256) void k_tmax(const float* __restrict__ tv,
                                              float* __restrict__ tmax)
{
    __shared__ float red[256];
    const int h = blockIdx.x;                 // 0..7 == b*4 + head
    float m = -1e30f;
    for (int j = threadIdx.x; j < NNODES; j += 256)
        m = fmaxf(m, tv[(size_t)h * NNODES + j]);
    red[threadIdx.x] = m;
    __syncthreads();
    for (int s = 128; s > 0; s >>= 1) {
        if (threadIdx.x < (unsigned)s)
            red[threadIdx.x] = fmaxf(red[threadIdx.x], red[threadIdx.x + s]);
        __syncthreads();
    }
    if (threadIdx.x == 0) tmax[h] = red[0];
}

__device__ __forceinline__ float prob_el(float sb, float Mi, float tj, int aij)
{
    float e = sb + tj;
    float l = fmaxf(e, NEG_SLOPE * e);                 // LeakyReLU(0.2)
    float p = __builtin_amdgcn_exp2f(fmaf(l, LOG2E, -Mi));
    return aij ? p : 0.0f;
}

// ---------------------------------------------------------------------------
// Kernel 2: flash-style masked-softmax aggregation + residual + LayerNorm.
// Block: 128 threads = 4 waves; wave w = head w; block owns 16 rows (i-tile).
// j loop in steps of 32 (= WMMA K):
//   - adj tile staged via double-buffered global_load_async_to_lds_b128
//     (ASYNCcnt-tracked CDNA5 async copy), shared by all 4 heads
//   - probabilities computed in the WMMA A-register layout
//   - 3 x v_wmma_f32_16x16x32_f16 per head per step: two for P@H (d tiles),
//     one against an all-ones B to produce the softmax row sums for free.
// ---------------------------------------------------------------------------
__global__ __launch_bounds__(128) void k_attn(
    const float* __restrict__ x, const int* __restrict__ adj,
    const float* __restrict__ att_bias,
    const float* __restrict__ ln_gamma, const float* __restrict__ ln_beta,
    const _Float16* __restrict__ hT, const float* __restrict__ sv,
    const float* __restrict__ tv, const float* __restrict__ tmax,
    float* __restrict__ out)
{
    __shared__ __align__(16) int   adjS[2][16 * 36]; // double-buffered, padded stride
    __shared__ __align__(16) float outS[16 * 132];

    const int tid  = threadIdx.x;
    const int lane = tid & 31;
    const int wave = tid >> 5;
    const int mrow = lane & 15;
    const int g    = lane >> 4;
    const int b    = blockIdx.y;
    const int i0   = blockIdx.x * 16;

    // Per-lane row constants (softmax upper bound M_i = LeakyReLU(s_i+Tmax+bias)*log2e)
    const float bias = att_bias[wave];
    const float sb   = sv[(size_t)(b * NHEADS + wave) * NNODES + i0 + mrow] + bias;
    const float tm   = tmax[b * NHEADS + wave];
    float lm = sb + tm;
    lm = fmaxf(lm, NEG_SLOPE * lm);
    const float Mi = lm * LOG2E;

    const float*    tB = tv + (size_t)(b * NHEADS + wave) * NNODES;
    const _Float16* h0 = hT + (size_t)(b * DMODEL + wave * HDIM + mrow) * NNODES;
    const _Float16* h1 = hT + (size_t)(b * DMODEL + wave * HDIM + 16 + mrow) * NNODES;
    const int* arow = adj + ((size_t)b * NNODES + i0) * NNODES;

    // async-staging addresses: each thread moves one 16B chunk of the 16x32 tile
    const int srr = tid >> 3;          // tile row 0..15
    const int scc = (tid & 7) * 4;     // 4-dword column group
    const unsigned ldsb0 = (unsigned)(uintptr_t)&adjS[0][srr * 36 + scc];
    const unsigned ldsb1 = (unsigned)(uintptr_t)&adjS[1][srr * 36 + scc];
    const int* gsrc = arow + (size_t)srr * NNODES + scc;

    // all-ones B fragment: D = P * ones puts each row's sum in every lane slot
    v16h ones;
#pragma unroll
    for (int e = 0; e < 16; ++e) ones[e] = (_Float16)1.0f;

    v8f acc0 = {}, acc1 = {}, accS = {};

    // preload tile 0 into buffer 0 (ASYNCcnt)
    asm volatile("global_load_async_to_lds_b128 %0, %1, off"
                 :: "v"(ldsb0), "v"((unsigned long long)(uintptr_t)gsrc)
                 : "memory");

    for (int j0 = 0; j0 < NNODES; j0 += 32) {
        const int cur = (j0 >> 5) & 1;
        // wait for this iteration's tile, then make it visible to all waves
        asm volatile("s_wait_asynccnt 0" ::: "memory");
        __syncthreads();
        // kick off next tile into the other buffer (its last readers finished
        // before the barrier above)
        if (j0 + 32 < NNODES) {
            asm volatile("global_load_async_to_lds_b128 %0, %1, off"
                         :: "v"(cur ? ldsb0 : ldsb1),
                            "v"((unsigned long long)(uintptr_t)(gsrc + j0 + 32))
                         : "memory");
        }

        // ---- probabilities in WMMA A layout: lane (g,mrow) covers row mrow,
        //      K = 16g..16g+15 ----
        const int* am = &adjS[cur][mrow * 36];
        int4 a0 = *reinterpret_cast<const int4*>(am + 16 * g);
        int4 a1 = *reinterpret_cast<const int4*>(am + 16 * g + 4);
        int4 a2 = *reinterpret_cast<const int4*>(am + 16 * g + 8);
        int4 a3 = *reinterpret_cast<const int4*>(am + 16 * g + 12);
        float4 t0 = *reinterpret_cast<const float4*>(tB + j0 + 16 * g);
        float4 t1 = *reinterpret_cast<const float4*>(tB + j0 + 16 * g + 4);
        float4 t2 = *reinterpret_cast<const float4*>(tB + j0 + 16 * g + 8);
        float4 t3 = *reinterpret_cast<const float4*>(tB + j0 + 16 * g + 12);

        v16h A;
        A[0]  = (_Float16)prob_el(sb, Mi, t0.x, a0.x);
        A[1]  = (_Float16)prob_el(sb, Mi, t0.y, a0.y);
        A[2]  = (_Float16)prob_el(sb, Mi, t0.z, a0.z);
        A[3]  = (_Float16)prob_el(sb, Mi, t0.w, a0.w);
        A[4]  = (_Float16)prob_el(sb, Mi, t1.x, a1.x);
        A[5]  = (_Float16)prob_el(sb, Mi, t1.y, a1.y);
        A[6]  = (_Float16)prob_el(sb, Mi, t1.z, a1.z);
        A[7]  = (_Float16)prob_el(sb, Mi, t1.w, a1.w);
        A[8]  = (_Float16)prob_el(sb, Mi, t2.x, a2.x);
        A[9]  = (_Float16)prob_el(sb, Mi, t2.y, a2.y);
        A[10] = (_Float16)prob_el(sb, Mi, t2.z, a2.z);
        A[11] = (_Float16)prob_el(sb, Mi, t2.w, a2.w);
        A[12] = (_Float16)prob_el(sb, Mi, t3.x, a3.x);
        A[13] = (_Float16)prob_el(sb, Mi, t3.y, a3.y);
        A[14] = (_Float16)prob_el(sb, Mi, t3.z, a3.z);
        A[15] = (_Float16)prob_el(sb, Mi, t3.w, a3.w);

        // ---- B fragments: contiguous 16 f16 along j from hT (32B aligned) ----
        const v16h B0 = *reinterpret_cast<const v16h*>(h0 + j0 + 16 * g);
        const v16h B1 = *reinterpret_cast<const v16h*>(h1 + j0 + 16 * g);

        acc0 = __builtin_amdgcn_wmma_f32_16x16x32_f16(false, A, false, B0,   (short)0, acc0, false, false);
        acc1 = __builtin_amdgcn_wmma_f32_16x16x32_f16(false, A, false, B1,   (short)0, acc1, false, false);
        accS = __builtin_amdgcn_wmma_f32_16x16x32_f16(false, A, false, ones, (short)0, accS, false, false);
    }

    // normalize and scatter to LDS out tile (row = 8g + r, col = head dims).
    // accS[r] holds the full row sum for row 8g+r in every lane (ones-B trick).
#pragma unroll
    for (int r = 0; r < 8; ++r) {
        float inv = 1.0f / fmaxf(accS[r], 1e-20f);
        outS[(8 * g + r) * 132 + wave * 32 + mrow]      = acc0[r] * inv;
        outS[(8 * g + r) * 132 + wave * 32 + 16 + mrow] = acc1[r] * inv;
    }
    __syncthreads();

    // ---- fused residual + LayerNorm: 8 threads per row, 16 dims each ----
    const int row = tid >> 3;
    const int c0  = (tid & 7) * 16;
    const float* xr = x + ((size_t)b * NNODES + i0 + row) * DMODEL + c0;
    float vals[16];
    float vsum = 0.0f;
#pragma unroll
    for (int k = 0; k < 16; k += 4) {
        float4 a = *reinterpret_cast<const float4*>(&outS[row * 132 + c0 + k]);
        float4 xv = *reinterpret_cast<const float4*>(xr + k);
        vals[k]     = a.x + xv.x; vals[k + 1] = a.y + xv.y;
        vals[k + 2] = a.z + xv.z; vals[k + 3] = a.w + xv.w;
        vsum += vals[k] + vals[k + 1] + vals[k + 2] + vals[k + 3];
    }
    vsum += __shfl_xor(vsum, 1, 32);
    vsum += __shfl_xor(vsum, 2, 32);
    vsum += __shfl_xor(vsum, 4, 32);
    const float mu = vsum * (1.0f / DMODEL);
    float vvar = 0.0f;
#pragma unroll
    for (int k = 0; k < 16; ++k) { float d = vals[k] - mu; vvar += d * d; }
    vvar += __shfl_xor(vvar, 1, 32);
    vvar += __shfl_xor(vvar, 2, 32);
    vvar += __shfl_xor(vvar, 4, 32);
    const float rstd = rsqrtf(vvar * (1.0f / DMODEL) + LN_EPS);

    float* op = out + ((size_t)b * NNODES + i0 + row) * DMODEL + c0;
#pragma unroll
    for (int k = 0; k < 16; k += 4) {
        float4 gm = *reinterpret_cast<const float4*>(ln_gamma + c0 + k);
        float4 bt = *reinterpret_cast<const float4*>(ln_beta + c0 + k);
        float4 o;
        o.x = (vals[k]     - mu) * rstd * gm.x + bt.x;
        o.y = (vals[k + 1] - mu) * rstd * gm.y + bt.y;
        o.z = (vals[k + 2] - mu) * rstd * gm.z + bt.z;
        o.w = (vals[k + 3] - mu) * rstd * gm.w + bt.w;
        *reinterpret_cast<float4*>(op + k) = o;
    }
}

// ---------------------------------------------------------------------------
extern "C" void kernel_launch(void* const* d_in, const int* in_sizes, int n_in,
                              void* d_out, int out_size, void* d_ws, size_t ws_size,
                              hipStream_t stream)
{
    const float* x        = (const float*)d_in[0];
    const int*   adj      = (const int*)d_in[1];
    const float* W        = (const float*)d_in[2];
    const float* att_src  = (const float*)d_in[3];
    const float* att_dst  = (const float*)d_in[4];
    const float* att_bias = (const float*)d_in[5];
    const float* ln_gamma = (const float*)d_in[6];
    const float* ln_beta  = (const float*)d_in[7];
    float* out = (float*)d_out;

    // workspace: hT (f16, 2MiB) | s (128KiB) | t (128KiB) | tmax (32B)
    _Float16* hT = (_Float16*)d_ws;
    float* sv   = (float*)((char*)d_ws + (size_t)NB * DMODEL * NNODES * sizeof(_Float16));
    float* tv   = sv + (size_t)NB * NHEADS * NNODES;
    float* tmax = tv + (size_t)NB * NHEADS * NNODES;

    k_proj<<<dim3(NB * NNODES / 16), dim3(128), 0, stream>>>(x, W, att_src, att_dst, hT, sv, tv);
    k_tmax<<<dim3(NB * NHEADS), dim3(256), 0, stream>>>(tv, tmax);
    k_attn<<<dim3(NNODES / 16, NB), dim3(128), 0, stream>>>(
        x, adj, att_bias, ln_gamma, ln_beta, hT, sv, tv, tmax, out);
}
